// GraphConvLayer_4346506903598
// MI455X (gfx1250) — compile-verified, matrix operand order checked by reading, not run on hardware
//
#include <hip/hip_runtime.h>
#include <hip/hip_bf16.h>

typedef _Float16 v16h __attribute__((ext_vector_type(16)));
typedef _Float16 v8h  __attribute__((ext_vector_type(8)));
typedef _Float16 v4h  __attribute__((ext_vector_type(4)));
typedef float    v8f  __attribute__((ext_vector_type(8)));
typedef float    v4f  __attribute__((ext_vector_type(4)));

#define F_DIM 128
#define BDIM  2

// ---------------- degree / normalization ----------------

__global__ void deg_init_kernel(float* __restrict__ deg, int N) {
    int i = blockIdx.x * blockDim.x + threadIdx.x;
    if (i < N) deg[i] = 1.0f;              // self-loop weight
}

__global__ void deg_accum_kernel(const long long* __restrict__ ei,
                                 const float* __restrict__ ea,
                                 float* __restrict__ deg, int E) {
    int e = blockIdx.x * blockDim.x + threadIdx.x;
    if (e < E) {
        int r = (int)ei[e];                // row = edge_index[0]
        atomicAdd(&deg[r], ea[e]);
    }
}

__global__ void dinv_kernel(const float* __restrict__ deg,
                            float* __restrict__ dinv, int N) {
    int i = blockIdx.x * blockDim.x + threadIdx.x;
    if (i < N) {
        float d = deg[i];
        dinv[i] = (d > 0.0f) ? rsqrtf(d) : 0.0f;
    }
}

// ---------------- pack W into WMMA B-operand layout (f16) ----------------
// Wpack index: (((kt*8 + t)*32 + lane)*16 + j)
//   n = t*16 + (lane&15);  k = kt*32 + (j<8?0:16) + (lane>>4)*8 + (j&7)
//   value = W[n*128 + k]   (B[k][n] = W[o=n][f=k])

__global__ void packw_kernel(const float* __restrict__ W,
                             _Float16* __restrict__ Wpack) {
    int i = blockIdx.x * blockDim.x + threadIdx.x;
    if (i >= 4 * 8 * 32 * 16) return;
    int j    = i & 15;
    int lane = (i >> 4) & 31;
    int t    = (i >> 9) & 7;
    int kt   = i >> 12;
    int n  = t * 16 + (lane & 15);
    int hi = lane >> 4;
    int k  = kt * 32 + ((j < 8) ? 0 : 16) + hi * 8 + (j & 7);
    Wpack[i] = (_Float16)W[n * F_DIM + k];
}

// ---------------- GEMM: x_lin[r, o] = sum_f x[r,f]*W[o,f] + b[o] ----------------
// 256 threads = 8 waves; wave w computes rows [blk*128 + w*16, +16) x all 128 outs.

__global__ __launch_bounds__(256) void gemm_kernel(
        const float* __restrict__ x, const _Float16* __restrict__ Wpack,
        const float* __restrict__ bias, float* __restrict__ xlin, int R) {
    __shared__ _Float16 Xs[128 * 32];      // 8 KB f16 staging tile

    const int tid  = threadIdx.x;
    const int wv   = tid >> 5;
    const int lane = tid & 31;
    const int m    = lane & 15;
    const int hi   = lane >> 4;
    const long long rowbase = (long long)blockIdx.x * 128;

    v8f acc[8] = {};

    for (int kt = 0; kt < 4; ++kt) {
        const int kc = kt * 32;

        // stage 128x32 f32 chunk of x -> f16 LDS.
        // Branchless: clamp row address, zero via value mask; float4 load
        // (global_load_b128) -> packed cvt -> 4-half store (ds_store_b64).
#pragma unroll
        for (int it = 0; it < 4; ++it) {
            int chunk = tid + 256 * it;            // 0..1023, each = 4 floats
            int r  = chunk >> 3;
            int c4 = (chunk & 7) * 4;
            long long gr = rowbase + r;
            long long gc = (gr < (long long)(R - 1)) ? gr : (long long)(R - 1);
            v4f v = *(const v4f*)&x[gc * F_DIM + kc + c4];
            float msk = (gr < R) ? 1.0f : 0.0f;
            v = v * msk;
            *(v4h*)&Xs[r * 32 + c4] = __builtin_convertvector(v, v4h);
        }
        __syncthreads();

        // Build A operand per ISA layout: lane m, K runs hi*8..+7 and 16+hi*8..+7
        const int arow = wv * 16 + m;
        v8h alo = *(const v8h*)&Xs[arow * 32 + hi * 8];
        v8h ahi = *(const v8h*)&Xs[arow * 32 + 16 + hi * 8];
        v16h a;
#pragma unroll
        for (int j = 0; j < 8; ++j) { a[j] = alo[j]; a[j + 8] = ahi[j]; }

        // prefetch next K chunk of x (global_prefetch_b8 path)
        if (kt < 3)
            __builtin_prefetch(&x[(rowbase + wv * 16 + m) * F_DIM + kc + 32], 0, 1);

#pragma unroll
        for (int t = 0; t < 8; ++t) {
            v16h bm = *(const v16h*)&Wpack[(((kt * 8 + t) * 32) + lane) * 16];
            acc[t] = __builtin_amdgcn_wmma_f32_16x16x32_f16(
                false, a, false, bm, (short)0, acc[t], false, false);
        }
        __syncthreads();
    }

    // C/D layout: VGPR r holds (M = r + hi*8, N = lane&15) of each 16x16 tile
#pragma unroll
    for (int t = 0; t < 8; ++t) {
        int n = t * 16 + m;
        float bv = bias[n];
#pragma unroll
        for (int r = 0; r < 8; ++r) {
            long long grow = rowbase + wv * 16 + hi * 8 + r;
            if (grow < R) xlin[grow * F_DIM + n] = acc[t][r] + bv;
        }
    }
}

// ---------------- message passing ----------------

// self-loop term: out = dinv[n]^2 * x_lin (also initializes d_out)
__global__ void out_init_kernel(const float* __restrict__ xlin,
                                const float* __restrict__ dinv,
                                float* __restrict__ out, int N) {
    long long idx = (long long)blockIdx.x * blockDim.x + threadIdx.x;
    long long total = (long long)BDIM * N * F_DIM;
    if (idx >= total) return;
    int n = (int)((idx >> 7) % N);
    float dv = dinv[n];
    out[idx] = dv * dv * xlin[idx];
}

// edge term: out[b, row, o] += dinv[row]*w*dinv[col] * x_lin[b, col, o]
__global__ void scatter_kernel(const long long* __restrict__ ei,
                               const float* __restrict__ ea,
                               const float* __restrict__ dinv,
                               const float* __restrict__ xlin,
                               float* __restrict__ out, int E, int N) {
    long long idx = (long long)blockIdx.x * blockDim.x + threadIdx.x;
    long long total = (long long)E * F_DIM;
    if (idx >= total) return;
    int e = (int)(idx >> 7);
    int o = (int)(idx & (F_DIM - 1));
    int r = (int)ei[e];
    int c = (int)ei[(long long)E + e];
    float nrm = dinv[r] * ea[e] * dinv[c];
    // batch 0 and 1 (x_lin / out laid out [B, N, F])
    atomicAdd(&out[(long long)r * F_DIM + o],
              nrm * xlin[(long long)c * F_DIM + o]);
    atomicAdd(&out[((long long)N + r) * F_DIM + o],
              nrm * xlin[((long long)N + c) * F_DIM + o]);
}

__global__ void relu_kernel(float* __restrict__ out, long long total) {
    long long idx = (long long)blockIdx.x * blockDim.x + threadIdx.x;
    if (idx < total) out[idx] = fmaxf(out[idx], 0.0f);
}

// ---------------- launch ----------------

extern "C" void kernel_launch(void* const* d_in, const int* in_sizes, int n_in,
                              void* d_out, int out_size, void* d_ws, size_t ws_size,
                              hipStream_t stream) {
    const float*      x    = (const float*)d_in[0];      // [B, N, 128] f32
    const long long*  ei   = (const long long*)d_in[1];  // [2, E] int64
    const float*      ea   = (const float*)d_in[2];      // [E, 1] f32
    const float*      W    = (const float*)d_in[3];      // [128, 128] f32
    const float*      bias = (const float*)d_in[4];      // [128] f32
    float*            out  = (float*)d_out;              // [B, N, 128] f32

    const int N = in_sizes[0] / (BDIM * F_DIM);          // 50000
    const int E = in_sizes[2];                           // 800000
    const int R = BDIM * N;                              // 100000 GEMM rows

    // workspace carve-out
    char* wsb = (char*)d_ws;
    float*    deg   = (float*)wsb;                                   // N f32
    float*    dinv  = deg + N;                                       // N f32
    _Float16* Wpack = (_Float16*)(wsb + (size_t)2 * N * sizeof(float)); // 32 KB
    float*    xlin  = (float*)(wsb + (size_t)2 * N * sizeof(float) + 32768); // R*128 f32

    const int T = 256;

    // 1-3: GCN normalization
    deg_init_kernel<<<(N + T - 1) / T, T, 0, stream>>>(deg, N);
    deg_accum_kernel<<<(E + T - 1) / T, T, 0, stream>>>(ei, ea, deg, E);
    dinv_kernel<<<(N + T - 1) / T, T, 0, stream>>>(deg, dinv, N);

    // 4: pre-swizzle W into WMMA B-operand layout (f16)
    packw_kernel<<<(4 * 8 * 32 * 16 + T - 1) / T, T, 0, stream>>>(W, Wpack);

    // 5: WMMA GEMM
    gemm_kernel<<<(R + 127) / 128, T, 0, stream>>>(x, Wpack, bias, xlin, R);

    // 6-8: message passing + ReLU
    long long outTotal = (long long)BDIM * N * F_DIM;
    out_init_kernel<<<(int)((outTotal + T - 1) / T), T, 0, stream>>>(xlin, dinv, out, N);
    long long edgeTotal = (long long)E * F_DIM;
    scatter_kernel<<<(int)((edgeTotal + T - 1) / T), T, 0, stream>>>(ei, ea, dinv, xlin, out, E, N);
    relu_kernel<<<(int)((outTotal + T - 1) / T), T, 0, stream>>>(out, outTotal);
}
